// SoftAttentionDrop_16621523436212
// MI455X (gfx1250) — matrix operand order repaired and checked by reading, not compiled, not run on hardware
//
#include <hip/hip_runtime.h>

typedef __attribute__((ext_vector_type(16))) _Float16 v16h;
typedef __attribute__((ext_vector_type(8)))  float    v8f;
typedef __attribute__((ext_vector_type(4)))  float    v4f;

#define DIM            256
#define ROWS_PER_BLOCK 16
#define NWAVES         4
#define TPW            4      // column tiles (16 cols each) per wave
#define KCHUNKS        8      // 256 / 32
#define PEEL_K         26
#define TEMP_INV       5.0f               // 1 / 0.2
#define OUT_SCALE      (1.0f / 0.9f)      // 1 / (1 - P)

// ---------------------------------------------------------------------------
// Pre-kernel: convert W (f32 [256 x 256], W[j][d]) into f16 WMMA B-fragments.
// Fragment (T,c) is the B matrix (32x16) for cols j in [16T,16T+16),
// k in [32c,32c+32).  B layout used here (16-bit, 32x16): lane l holds col j=l&15,
// K = (l>>4)*16 + {2v,2v+1} in VGPR v  ->  16 contiguous k values per lane.
// ---------------------------------------------------------------------------
__global__ void wconv_kernel(const float* __restrict__ W,
                             _Float16* __restrict__ Wh) {
    const int frag = blockIdx.x;            // 0..127  (= T*8 + c)
    const int l    = threadIdx.x;           // 0..31
    const int T    = frag >> 3;
    const int c    = frag & 7;
    const int j    = T * 16 + (l & 15);
    const int k    = c * 32 + (l >> 4) * 16;
    const float* src = W + j * DIM + k;
    v16h h;
#pragma unroll
    for (int i = 0; i < 16; ++i) h[i] = (_Float16)src[i];
    *(v16h*)(Wh + (size_t)(frag * 32 + l) * 16) = h;
}

// ---------------------------------------------------------------------------
// Fused kernel: 16-row block per workgroup (4 wave32).  GEMM via
// v_wmma_f32_16x16x32_f16, then 26 softmax-peel iterations entirely in
// registers (shuffle + tiny LDS reductions), then scaled output.
// ---------------------------------------------------------------------------
__launch_bounds__(NWAVES * 32)
__global__ void sad_fused_kernel(const float* __restrict__ feature,
                                 const _Float16* __restrict__ Wh,
                                 const float* __restrict__ bias,
                                 float* __restrict__ out) {
    __shared__ float red[ROWS_PER_BLOCK * NWAVES];   // [row][wave]

    const int tid  = threadIdx.x;
    const int wv   = tid >> 5;          // wave in workgroup
    const int l    = tid & 31;          // lane
    const int lmod = l & 15;
    const int lhi  = l >> 4;
    const long row0 = (long)blockIdx.x * ROWS_PER_BLOCK;

    // ---------------- GEMM: mask = feature @ W^T + b --------------------
    // C/D layout: lane l, reg r  ->  row m = r + 8*lhi, col j = tile*16 + lmod
    v8f acc[TPW] = {};

    // A layout (16-bit A 16x32): lane l = row l&15; contiguous run
    // [16*lhi, 16*lhi+16) of the 32-wide k chunk.
    const float* arow = feature + (row0 + lmod) * DIM + lhi * 16;
#pragma unroll
    for (int c = 0; c < KCHUNKS; ++c) {
        v4f a0 = *(const v4f*)(arow + c * 32);
        v4f a1 = *(const v4f*)(arow + c * 32 + 4);
        v4f a2 = *(const v4f*)(arow + c * 32 + 8);
        v4f a3 = *(const v4f*)(arow + c * 32 + 12);
        v16h a;
#pragma unroll
        for (int i = 0; i < 4; ++i) {
            a[i]      = (_Float16)a0[i];
            a[i + 4]  = (_Float16)a1[i];
            a[i + 8]  = (_Float16)a2[i];
            a[i + 12] = (_Float16)a3[i];
        }
#pragma unroll
        for (int t = 0; t < TPW; ++t) {
            const int frag = (wv * TPW + t) * KCHUNKS + c;
            v16h bm = *(const v16h*)(Wh + (size_t)(frag * 32 + l) * 16);
            acc[t] = __builtin_amdgcn_wmma_f32_16x16x32_f16(
                false, a, false, bm, (short)0, acc[t], false, false);
        }
    }
    // bias: col j is constant over the 8 regs of a tile
#pragma unroll
    for (int t = 0; t < TPW; ++t) {
        const float bj = bias[(wv * TPW + t) * 16 + lmod];
#pragma unroll
        for (int r = 0; r < 8; ++r) acc[t][r] += bj;
    }

    // ---------------- 26 softmax-peel iterations ------------------------
    v8f y[TPW] = {};
    v8f p[TPW];

    for (int it = 0; it < PEEL_K; ++it) {
        // logits = (mask + log(1-y+eps)) / TEMP ; track per-row max
        float rmax[8];
#pragma unroll
        for (int r = 0; r < 8; ++r) rmax[r] = -3.4e38f;
#pragma unroll
        for (int t = 0; t < TPW; ++t)
#pragma unroll
            for (int r = 0; r < 8; ++r) {
                const float w  = 1.0f - y[t][r];
                const float lg = (acc[t][r] + __logf(w + 1e-12f)) * TEMP_INV;
                p[t][r] = lg;
                rmax[r] = fmaxf(rmax[r], lg);
            }
        // reduce across the 16 lanes holding this row (xor 1,2,4,8 stays
        // inside each half-wave, halves hold disjoint rows)
#pragma unroll
        for (int r = 0; r < 8; ++r) {
#pragma unroll
            for (int d = 1; d < 16; d <<= 1)
                rmax[r] = fmaxf(rmax[r], __shfl_xor(rmax[r], d, 32));
        }
        // combine across the 4 waves via LDS
        __syncthreads();
        if (lmod == 0) {
#pragma unroll
            for (int r = 0; r < 8; ++r)
                red[(r + 8 * lhi) * NWAVES + wv] = rmax[r];
        }
        __syncthreads();
#pragma unroll
        for (int r = 0; r < 8; ++r) {
            v4f m4 = *(const v4f*)&red[(r + 8 * lhi) * NWAVES];
            rmax[r] = fmaxf(fmaxf(m4.x, m4.y), fmaxf(m4.z, m4.w));
        }

        // exp + per-row sum
        float rsum[8];
#pragma unroll
        for (int r = 0; r < 8; ++r) rsum[r] = 0.0f;
#pragma unroll
        for (int t = 0; t < TPW; ++t)
#pragma unroll
            for (int r = 0; r < 8; ++r) {
                const float e = __expf(p[t][r] - rmax[r]);
                p[t][r] = e;
                rsum[r] += e;
            }
#pragma unroll
        for (int r = 0; r < 8; ++r) {
#pragma unroll
            for (int d = 1; d < 16; d <<= 1)
                rsum[r] += __shfl_xor(rsum[r], d, 32);
        }
        __syncthreads();
        if (lmod == 0) {
#pragma unroll
            for (int r = 0; r < 8; ++r)
                red[(r + 8 * lhi) * NWAVES + wv] = rsum[r];
        }
        __syncthreads();
#pragma unroll
        for (int r = 0; r < 8; ++r) {
            v4f s4 = *(const v4f*)&red[(r + 8 * lhi) * NWAVES];
            rsum[r] = 1.0f / (s4.x + s4.y + s4.z + s4.w);   // reciprocal of row sum
        }

        // y += softmax * w
#pragma unroll
        for (int t = 0; t < TPW; ++t)
#pragma unroll
            for (int r = 0; r < 8; ++r) {
                const float w = 1.0f - y[t][r];
                y[t][r] += p[t][r] * rsum[r] * w;
            }
    }

    // ---------------- out = feature * (1-y) / (1-P) ---------------------
#pragma unroll
    for (int t = 0; t < TPW; ++t) {
        const int j = (wv * TPW + t) * 16 + lmod;
#pragma unroll
        for (int r = 0; r < 8; ++r) {
            const long m = row0 + r + 8 * lhi;
            const float f = feature[m * DIM + j];
            out[m * DIM + j] = f * (1.0f - y[t][r]) * OUT_SCALE;
        }
    }
}

// ---------------------------------------------------------------------------
extern "C" void kernel_launch(void* const* d_in, const int* in_sizes, int n_in,
                              void* d_out, int out_size, void* d_ws, size_t ws_size,
                              hipStream_t stream) {
    const float* feature = (const float*)d_in[0];   // [N, 256] f32
    const float* W       = (const float*)d_in[1];   // [256, 256] f32
    const float* b       = (const float*)d_in[2];   // [256] f32
    float* out           = (float*)d_out;           // [N, 256] f32
    _Float16* Wh         = (_Float16*)d_ws;         // 128 KB of fragments

    // Re-lay W into f16 WMMA B-fragments (deterministic, every call).
    wconv_kernel<<<128, 32, 0, stream>>>(W, Wh);

    const int nrows   = in_sizes[0] / DIM;            // 300000
    const int nblocks = nrows / ROWS_PER_BLOCK;       // 18750 (exact)
    sad_fused_kernel<<<nblocks, NWAVES * 32, 0, stream>>>(feature, Wh, b, out);
}
